// DualAttention_79869211836735
// MI455X (gfx1250) — matrix operand order repaired and checked by reading, not compiled, not run on hardware
//
#include <hip/hip_runtime.h>
#include <hip/hip_bf16.h>
#include <math.h>

// ---------------------------------------------------------------------------
// Types / WMMA helpers (CDNA5 gfx1250, wave32)
// ---------------------------------------------------------------------------
typedef _Float16 half_t;
typedef __attribute__((ext_vector_type(16))) _Float16 v16h;
typedef __attribute__((ext_vector_type(8)))  float    v8f;

static __device__ __forceinline__ v8f wmma_f16(v16h a, v16h b, v8f c) {
  // D = A(16x32 f16) * B(32x16 f16) + C(16x16 f32)
  return __builtin_amdgcn_wmma_f32_16x16x32_f16(false, a, false, b, (short)0, c,
                                                false, false);
}

// A-fragment: 16x32 f16 tile, row-major LDS, leading dim `ld` (halves).
// ISA 7.12.2: row = lane&15; lanes>=16 add +8 to k; VGPR j holds
// k = (j>>2)*16 + (j&3)*2 (+1).  Two contiguous 16B runs -> 2x ds_load_b128.
static __device__ __forceinline__ v16h load_frag_a(const half_t* s, int row0,
                                                   int ld, int kbase, int lane) {
  v16h a;
  const int r  = row0 + (lane & 15);
  const int kb = kbase + ((lane >> 4) << 3);
#pragma unroll
  for (int j = 0; j < 8; ++j) {
    const int k = kb + ((j >> 2) << 4) + ((j & 3) << 1);
    a[2 * j]     = s[r * ld + k];
    a[2 * j + 1] = s[r * ld + k + 1];
  }
  return a;
}

// B-fragment from an N-major (transposed) LDS tile S[N][K], leading dim `ld`:
// B[k][n] = S[n*ld + k].  ISA: col = lane&15; lanes>=16 add +8 to k;
// VGPR j holds k = (j>>2)*16 + (j&3)*2 (+1).  Two contiguous 16B runs -> 2x ds_load_b128.
static __device__ __forceinline__ v16h load_frag_bt(const half_t* s, int col0,
                                                    int ld, int kbase, int lane) {
  v16h b;
  const int c  = col0 + (lane & 15);
  const int kb = kbase + ((lane >> 4) << 3);
#pragma unroll
  for (int j = 0; j < 8; ++j) {
    const int k = kb + ((j >> 2) << 4) + ((j & 3) << 1);
    b[2 * j]     = s[c * ld + k];
    b[2 * j + 1] = s[c * ld + k + 1];
  }
  return b;
}

// 4-element global loaders (16B for f32, 8B for f16), convert to float.
static __device__ __forceinline__ void load4(const float* p, float o[4]) {
  const float4 v = *(const float4*)p;
  o[0] = v.x; o[1] = v.y; o[2] = v.z; o[3] = v.w;
}
static __device__ __forceinline__ void load4(const half_t* p, float o[4]) {
  union { uint2 u; half_t h[4]; } t;
  t.u = *(const uint2*)p;
  o[0] = (float)t.h[0]; o[1] = (float)t.h[1];
  o[2] = (float)t.h[2]; o[3] = (float)t.h[3];
}

// ---------------------------------------------------------------------------
// Generic WMMA GEMM: C[M,N] = act(A[M,K] * B[K,N] + bias) (+ ewa.*ewb)
// Block tile 128x128, BK=32, 256 threads = 8 waves (2x4), wave tile 64x32
// (4x2 WMMA -> 8 WMMA per 6 fragment loads per BK step).
// A f32 or f16 (converted to f16 in LDS). B staged TRANSPOSED in LDS so all
// fragment reads are contiguous ds_load_b128. Optional per-(batch,k) scale on
// A (batch = row>>10) applied in the staging pass.
// LDS leading dims padded to 48 halves (96B: 16B-aligned, not 256B-periodic).
// ---------------------------------------------------------------------------
#define ACT_NONE 0
#define ACT_RELU 1
#define ACT_SIG  2
#define LDG 48

template <typename AT>
__global__ __launch_bounds__(256) void gemm_ws_kernel(
    const AT* __restrict__ A, const float* __restrict__ Bw,
    const float* __restrict__ bias, float* __restrict__ out32,
    half_t* __restrict__ out16, int M, int N, int K, int act,
    const float* __restrict__ a_scale, const float* __restrict__ ewa,
    const float* __restrict__ ewb) {
  __shared__ half_t sA[128 * LDG];   // [row][k]
  __shared__ half_t sBt[128 * LDG];  // [n][k]  (transposed B tile)

  const int tid  = threadIdx.x;
  const int lane = tid & 31;
  const int wid  = tid >> 5;
  const int wm   = wid >> 2;  // 0..1 (64 rows each)
  const int wn   = wid & 3;   // 0..3 (32 cols each)
  const int m0   = blockIdx.x * 128;
  const int n0   = blockIdx.y * 128;
  const int hi   = lane >> 4;
  const int c16  = lane & 15;

  v8f acc[4][2];
  const v8f vzero = {0.f, 0.f, 0.f, 0.f, 0.f, 0.f, 0.f, 0.f};
#pragma unroll
  for (int mt = 0; mt < 4; ++mt)
#pragma unroll
    for (int nt = 0; nt < 2; ++nt) acc[mt][nt] = vzero;

  for (int k0 = 0; k0 < K; k0 += 32) {
    __syncthreads();  // previous iteration's LDS reads complete
    // Stage A tile 128x32 -> sA (f16), 4-wide vector global loads.
    for (int i4 = tid; i4 < 1024; i4 += 256) {
      const int r = i4 >> 3, c4 = i4 & 7;  // 8 chunks of 4 per row
      float v[4];
      load4(&A[(size_t)(m0 + r) * K + k0 + c4 * 4], v);
      if (a_scale) {
        const float* sc = &a_scale[(size_t)((m0 + r) >> 10) * K + k0 + c4 * 4];
#pragma unroll
        for (int e = 0; e < 4; ++e) v[e] *= sc[e];
      }
#pragma unroll
      for (int e = 0; e < 4; ++e) sA[r * LDG + c4 * 4 + e] = (half_t)v[e];
    }
    // Stage B tile 32x128 -> sBt transposed (f16).
    for (int i4 = tid; i4 < 1024; i4 += 256) {
      const int r = i4 >> 5, c4 = i4 & 31;  // r = k row, 32 chunks of 4 cols
      float v[4];
      load4(&Bw[(size_t)(k0 + r) * N + n0 + c4 * 4], v);
#pragma unroll
      for (int e = 0; e < 4; ++e) sBt[(c4 * 4 + e) * LDG + r] = (half_t)v[e];
    }
    // Prefetch next A tile rows (gfx1250 global_prefetch path)
    if (k0 + 32 < K) {
      const int r = tid >> 1;
      __builtin_prefetch(&A[(size_t)(m0 + r) * K + k0 + 32], 0, 1);
    }
    __syncthreads();

    v16h af[4], bf[2];
#pragma unroll
    for (int mt = 0; mt < 4; ++mt)
      af[mt] = load_frag_a(sA, wm * 64 + mt * 16, LDG, 0, lane);
#pragma unroll
    for (int nt = 0; nt < 2; ++nt)
      bf[nt] = load_frag_bt(sBt, wn * 32 + nt * 16, LDG, 0, lane);
#pragma unroll
    for (int mt = 0; mt < 4; ++mt)
#pragma unroll
      for (int nt = 0; nt < 2; ++nt)
        acc[mt][nt] = wmma_f16(af[mt], bf[nt], acc[mt][nt]);
  }

  // Epilogue. C layout: VGPR j -> row = j + 8*hi, col = lane&15.
#pragma unroll
  for (int mt = 0; mt < 4; ++mt) {
#pragma unroll
    for (int nt = 0; nt < 2; ++nt) {
      const int gc      = n0 + wn * 32 + nt * 16 + c16;
      const int gr_base = m0 + wm * 64 + mt * 16 + hi * 8;
      const float bia   = bias ? bias[gc] : 0.f;
#pragma unroll
      for (int j = 0; j < 8; ++j) {
        const size_t idx = (size_t)(gr_base + j) * N + gc;
        float x = acc[mt][nt][j] + bia;
        if (act == ACT_RELU) x = fmaxf(x, 0.f);
        else if (act == ACT_SIG) x = 1.f / (1.f + __expf(-x));
        if (ewa) x += ewa[idx] * ewb[idx];
        if (out32) out32[idx] = x;
        if (out16) out16[idx] = (half_t)x;
      }
    }
  }
}

// ---------------------------------------------------------------------------
// Mean profile over sequence: prof[b,d] = mean_n value[b,n,d]
// ---------------------------------------------------------------------------
__global__ __launch_bounds__(256) void prof_kernel(const float* __restrict__ value,
                                                   float* __restrict__ prof) {
  const int idx = blockIdx.x * 256 + threadIdx.x;  // 0..4095
  const int b = idx >> 9, d = idx & 511;
  const float* p = value + (size_t)b * 1024 * 512 + d;
  float s = 0.f;
  for (int n = 0; n < 1024; ++n) s += p[(size_t)n * 512];
  prof[idx] = s * (1.f / 1024.f);
}

// ---------------------------------------------------------------------------
// Channel gate MLP (tiny): cw[b,:] = sigmoid(relu(prof[b]*Wc1+bc1)*Wc2+bc2)
// ---------------------------------------------------------------------------
__global__ __launch_bounds__(256) void cmlp_kernel(
    const float* __restrict__ prof, const float* __restrict__ Wc1,
    const float* __restrict__ bc1, const float* __restrict__ Wc2,
    const float* __restrict__ bc2, float* __restrict__ cw) {
  __shared__ float sp[512];
  __shared__ float sh[256];
  const int b = blockIdx.x, t = threadIdx.x;
  sp[t]       = prof[b * 512 + t];
  sp[t + 256] = prof[b * 512 + 256 + t];
  __syncthreads();
  float s = bc1[t];
  for (int d = 0; d < 512; ++d) s += sp[d] * Wc1[(size_t)d * 256 + t];
  sh[t] = fmaxf(s, 0.f);
  __syncthreads();
  for (int d = t; d < 512; d += 256) {
    float s2 = bc2[d];
    for (int j = 0; j < 256; ++j) s2 += sh[j] * Wc2[(size_t)j * 512 + d];
    cw[b * 512 + d] = 1.f / (1.f + __expf(-s2));
  }
}

// ---------------------------------------------------------------------------
// Flash-style attention: one block = 64 query rows of one (b, h).
// Q/K/Vdual in f16 merged layout [b*1024, 512], head h at cols h*64..h*64+63.
// 128 threads = 4 waves; wave = 16 query rows. Online softmax.
// V is staged TRANSPOSED in LDS so P@V B-fragments are contiguous b128 reads.
// LDS leading dim padded to 72 halves (144B).
// ---------------------------------------------------------------------------
#define LDA 72

__global__ __launch_bounds__(128) void attn_kernel(
    const half_t* __restrict__ Q, const half_t* __restrict__ Kh,
    const half_t* __restrict__ V, half_t* __restrict__ O) {
  __shared__ half_t sQ[64 * LDA];   // [qrow][dh]
  __shared__ half_t sK[64 * LDA];   // [key][dh]
  __shared__ half_t sVt[64 * LDA];  // [dh][key]  (transposed V tile)
  __shared__ half_t sP[64 * LDA];   // [qrow][key]

  const int tid  = threadIdx.x;
  const int lane = tid & 31;
  const int wid  = tid >> 5;  // 0..3
  const int hi   = lane >> 4;
  const int c16  = lane & 15;

  const int b = blockIdx.z, h = blockIdx.y, qt = blockIdx.x;
  const size_t tok0 = (size_t)b * 1024 + qt * 64;
  const int c0 = h * 64;

  // Stage Q tile (64x64), 4-half vector loads/stores.
  for (int i = tid; i < 1024; i += 128) {
    const int r = i >> 4, c4 = i & 15;
    *(uint2*)&sQ[r * LDA + c4 * 4] =
        *(const uint2*)&Q[(tok0 + r) * 512 + c0 + c4 * 4];
  }
  __syncthreads();

  v16h aq[2];
  aq[0] = load_frag_a(sQ, wid * 16, LDA, 0, lane);
  aq[1] = load_frag_a(sQ, wid * 16, LDA, 32, lane);

  const v8f vzero = {0.f, 0.f, 0.f, 0.f, 0.f, 0.f, 0.f, 0.f};
  v8f o[4];
  float mrun[8], lrun[8];
#pragma unroll
  for (int nt = 0; nt < 4; ++nt) o[nt] = vzero;
#pragma unroll
  for (int j = 0; j < 8; ++j) { mrun[j] = -3.0e38f; lrun[j] = 0.f; }

  for (int kt = 0; kt < 16; ++kt) {
    __syncthreads();  // previous P@V reads of sK/sVt done
    const size_t kb0 = (size_t)b * 1024 + kt * 64;
    for (int i = tid; i < 1024; i += 128) {
      const int r = i >> 4, c4 = i & 15;  // r = key, c4 = dh chunk
      *(uint2*)&sK[r * LDA + c4 * 4] =
          *(const uint2*)&Kh[(kb0 + r) * 512 + c0 + c4 * 4];
      union { uint2 u; half_t hh[4]; } t;
      t.u = *(const uint2*)&V[(kb0 + r) * 512 + c0 + c4 * 4];
#pragma unroll
      for (int e = 0; e < 4; ++e) sVt[(c4 * 4 + e) * LDA + r] = t.hh[e];
    }
    __syncthreads();

    // S = (Q * K^T) / 8 : 4 key n-tiles, 2 k-steps over Dh=64
    v8f s[4];
#pragma unroll
    for (int nt = 0; nt < 4; ++nt) {
      s[nt] = vzero;
#pragma unroll
      for (int ks = 0; ks < 2; ++ks) {
        v16h bf = load_frag_bt(sK, nt * 16, LDA, ks * 32, lane);
        s[nt] = wmma_f16(aq[ks], bf, s[nt]);
      }
    }
#pragma unroll
    for (int nt = 0; nt < 4; ++nt)
#pragma unroll
      for (int j = 0; j < 8; ++j) s[nt][j] *= 0.125f;

    // Online softmax per row (row = wid*16 + hi*8 + j; cols across 16 lanes)
#pragma unroll
    for (int j = 0; j < 8; ++j) {
      float rm = -3.0e38f;
#pragma unroll
      for (int nt = 0; nt < 4; ++nt) rm = fmaxf(rm, s[nt][j]);
      rm = fmaxf(rm, __shfl_xor(rm, 1));
      rm = fmaxf(rm, __shfl_xor(rm, 2));
      rm = fmaxf(rm, __shfl_xor(rm, 4));
      rm = fmaxf(rm, __shfl_xor(rm, 8));
      const float newm = fmaxf(mrun[j], rm);
      const float fac  = __expf(mrun[j] - newm);
      float rowsum = 0.f;
      const int prow = wid * 16 + hi * 8 + j;
#pragma unroll
      for (int nt = 0; nt < 4; ++nt) {
        const float p = __expf(s[nt][j] - newm);
        rowsum += p;
        sP[prow * LDA + nt * 16 + c16] = (half_t)p;
      }
      rowsum += __shfl_xor(rowsum, 1);
      rowsum += __shfl_xor(rowsum, 2);
      rowsum += __shfl_xor(rowsum, 4);
      rowsum += __shfl_xor(rowsum, 8);
      lrun[j] = lrun[j] * fac + rowsum;
      mrun[j] = newm;
#pragma unroll
      for (int nt = 0; nt < 4; ++nt) o[nt][j] *= fac;
    }
    __syncthreads();  // sP visible

    // O += P * V : 2 k-steps over 64 keys; B-fragments from transposed V.
#pragma unroll
    for (int ks = 0; ks < 2; ++ks) {
      v16h ap = load_frag_a(sP, wid * 16, LDA, ks * 32, lane);
#pragma unroll
      for (int nt = 0; nt < 4; ++nt) {
        v16h bf = load_frag_bt(sVt, nt * 16, LDA, ks * 32, lane);
        o[nt] = wmma_f16(ap, bf, o[nt]);
      }
    }
  }

  // Normalize and store merged-head f16 output
#pragma unroll
  for (int j = 0; j < 8; ++j) {
    const float inv = 1.f / lrun[j];
    const size_t row = tok0 + wid * 16 + hi * 8 + j;
#pragma unroll
    for (int nt = 0; nt < 4; ++nt)
      O[row * 512 + c0 + nt * 16 + c16] = (half_t)(o[nt][j] * inv);
  }
}

// ---------------------------------------------------------------------------
// Host-side orchestration
// ---------------------------------------------------------------------------
extern "C" void kernel_launch(void* const* d_in, const int* in_sizes, int n_in,
                              void* d_out, int out_size, void* d_ws,
                              size_t ws_size, hipStream_t stream) {
  (void)in_sizes; (void)n_in; (void)out_size; (void)ws_size;
  const float* query  = (const float*)d_in[0];
  const float* key_in = (const float*)d_in[1];
  const float* value  = (const float*)d_in[2];
  const float* Wq  = (const float*)d_in[3];  const float* bq  = (const float*)d_in[4];
  const float* Wk  = (const float*)d_in[5];  const float* bk  = (const float*)d_in[6];
  const float* Wv  = (const float*)d_in[7];  const float* bv  = (const float*)d_in[8];
  const float* Wo  = (const float*)d_in[9];  const float* bo  = (const float*)d_in[10];
  const float* Ws1 = (const float*)d_in[11]; const float* bs1 = (const float*)d_in[12];
  const float* Ws2 = (const float*)d_in[13]; const float* bs2 = (const float*)d_in[14];
  const float* Wc1 = (const float*)d_in[15]; const float* bc1 = (const float*)d_in[16];
  const float* Wc2 = (const float*)d_in[17]; const float* bc2 = (const float*)d_in[18];
  float* out = (float*)d_out;

  const int M = 8192;   // B*N tokens
  const int D = 512;
  const int Hm = 256;

  // Workspace carve-up (~68 MB)
  char* ws = (char*)d_ws;
  half_t* Qh    = (half_t*)ws;            ws += (size_t)M * D * sizeof(half_t);
  half_t* Khb   = (half_t*)ws;            ws += (size_t)M * D * sizeof(half_t);
  float*  VV    = (float*)ws;             ws += (size_t)M * D * sizeof(float);
  half_t* Hs    = (half_t*)ws;            ws += (size_t)M * Hm * sizeof(half_t);
  float*  SW    = (float*)ws;             ws += (size_t)M * D * sizeof(float);
  half_t* Vdual = (half_t*)ws;            ws += (size_t)M * D * sizeof(half_t);
  half_t* AttO  = (half_t*)ws;            ws += (size_t)M * D * sizeof(half_t);
  float*  prof  = (float*)ws;             ws += (size_t)8 * D * sizeof(float);
  float*  cw    = (float*)ws;             ws += (size_t)8 * D * sizeof(float);

  const dim3 blk(256);
  const dim3 gD(M / 128, D / 128);   // N = 512 -> (64, 4)
  const dim3 gH(M / 128, Hm / 128);  // N = 256 -> (64, 2)

  // Q = query*Wq + bq  -> f16
  gemm_ws_kernel<float><<<gD, blk, 0, stream>>>(query, Wq, bq, nullptr, Qh,
      M, D, D, ACT_NONE, nullptr, nullptr, nullptr);
  // K = key_in*Wk + bk -> f16
  gemm_ws_kernel<float><<<gD, blk, 0, stream>>>(key_in, Wk, bk, nullptr, Khb,
      M, D, D, ACT_NONE, nullptr, nullptr, nullptr);
  // VV = value*Wv + bv -> f32
  gemm_ws_kernel<float><<<gD, blk, 0, stream>>>(value, Wv, bv, VV, nullptr,
      M, D, D, ACT_NONE, nullptr, nullptr, nullptr);

  // prof[b,d] = mean_n value ; cw = channel gate
  prof_kernel<<<16, blk, 0, stream>>>(value, prof);
  cmlp_kernel<<<8, blk, 0, stream>>>(prof, Wc1, bc1, Wc2, bc2, cw);

  // Hs = relu(VV*Ws1 + bs1) -> f16
  gemm_ws_kernel<float><<<gH, blk, 0, stream>>>(VV, Ws1, bs1, nullptr, Hs,
      M, Hm, D, ACT_RELU, nullptr, nullptr, nullptr);
  // SW = sigmoid(Hs*Ws2 + bs2) -> f32
  gemm_ws_kernel<half_t><<<gD, blk, 0, stream>>>(Hs, Ws2, bs2, SW, nullptr,
      M, D, Hm, ACT_SIG, nullptr, nullptr, nullptr);

  // Vdual = (value .* cw[b,:])*Wv + bv  +  VV .* SW   -> f16
  gemm_ws_kernel<float><<<gD, blk, 0, stream>>>(value, Wv, bv, nullptr, Vdual,
      M, D, D, ACT_NONE, cw, VV, SW);

  // Attention: softmax(Q K^T / 8) * Vdual -> f16 merged-head layout
  attn_kernel<<<dim3(16, 8, 8), dim3(128), 0, stream>>>(Qh, Khb, Vdual, AttO);

  // out = AttO*Wo + bo -> f32
  gemm_ws_kernel<half_t><<<gD, blk, 0, stream>>>(AttO, Wo, bo, out, nullptr,
      M, D, D, ACT_NONE, nullptr, nullptr, nullptr);
}
